// SAE_16114717294669
// MI455X (gfx1250) — matrix operand order repaired and checked by reading, not compiled, not run on hardware
//
#include <hip/hip_runtime.h>
#include <stdint.h>

// Problem constants (from reference): B=4, D=16, L=256, H=512, M=4096, TOPK=64
#define H_     512
#define M_     4096
#define TOPK_  64
#define NROWS  16384   // B*D*L

typedef __attribute__((ext_vector_type(16))) __bf16 v16bf;
typedef __attribute__((ext_vector_type(8)))  __bf16 v8bf;
typedef __attribute__((ext_vector_type(8)))  float  v8f;
typedef __attribute__((ext_vector_type(4)))  unsigned int u32x4;
typedef __attribute__((ext_vector_type(8)))  int i32x8;
typedef __attribute__((ext_vector_type(4)))  int i32x4;

__device__ __forceinline__ __bf16 f32_to_bf16(float f) {
  uint32_t u = __float_as_uint(f);
  u += 0x7FFFu + ((u >> 16) & 1u);          // round-to-nearest-even
  uint16_t h = (uint16_t)(u >> 16);
  __bf16 r;
  __builtin_memcpy(&r, &h, sizeof(r));
  return r;
}

// ---------------------------------------------------------------------------
// Kernel 1: transpose dec [H][M] -> dec_t [M][H] so decode gathers are
// contiguous 2KB rows (coalesced, L2-resident).
// ---------------------------------------------------------------------------
__global__ __launch_bounds__(256)
void sae_transpose_dec(const float* __restrict__ dec, float* __restrict__ dect) {
  __shared__ float tile[32][33];
  const int m0 = blockIdx.x * 32;
  const int h0 = blockIdx.y * 32;
  const int tx = threadIdx.x, ty = threadIdx.y;   // 32 x 8
#pragma unroll
  for (int i = 0; i < 32; i += 8)
    tile[ty + i][tx] = dec[(size_t)(h0 + ty + i) * M_ + (m0 + tx)];
  __syncthreads();
#pragma unroll
  for (int i = 0; i < 32; i += 8)
    dect[(size_t)(m0 + ty + i) * H_ + (h0 + tx)] = tile[tx][ty + i];
}

// ---------------------------------------------------------------------------
// Kernel 2: convert encoder dictionary to bf16 once (saves per-WG re-convert).
// ---------------------------------------------------------------------------
__global__ __launch_bounds__(256)
void sae_convert_enc(const float* __restrict__ enc, __bf16* __restrict__ encb) {
  size_t i = (size_t)blockIdx.x * 256 + threadIdx.x;   // grid covers M_*H_
  encb[i] = f32_to_bf16(enc[i]);
}

// ---------------------------------------------------------------------------
// Kernel 3: fused encode GEMM (WMMA bf16 -> f32) + ReLU + exact top-64 radix
// select, logits tile fully LDS-resident (CDNA5 320KB LDS/WGP).
// A-tile staged global->LDS by the Tensor Data Mover (TENSORcnt).
// ---------------------------------------------------------------------------
#define A_PITCH 520          // 512 + 8 bf16 pad -> 16B aligned rows, no bank conflicts
#define L_PITCH 4112         // 4096 + 16 f32 pad
#define SMEM_A_BYTES (16 * A_PITCH * 2)        // 16,640
#define SMEM_L_BYTES (16 * L_PITCH * 4)        // 263,168
#define SMEM_H_BYTES (8 * 256 * 4)             // 8,192 (per-wave histograms)
#define SMEM_BYTES   (SMEM_A_BYTES + SMEM_L_BYTES + SMEM_H_BYTES + 64) // 288,064 < 320K

__global__ __launch_bounds__(256)
void sae_encode_topk(const float* __restrict__ x,
                     const __bf16* __restrict__ encb,
                     const float* __restrict__ bias_pre,
                     const float* __restrict__ bias_enc,
                     float* __restrict__ z_out,
                     float* __restrict__ topv,
                     int*   __restrict__ topi) {
  extern __shared__ char smem[];
  __bf16*   As   = (__bf16*)smem;
  float*    Llog = (float*)(smem + SMEM_A_BYTES);
  unsigned* Hist = (unsigned*)(smem + SMEM_A_BYTES + SMEM_L_BYTES);
  int*      Cnt  = (int*)(smem + SMEM_A_BYTES + SMEM_L_BYTES + SMEM_H_BYTES);

  const int tid  = threadIdx.x;
  const int w    = tid >> 5;       // wave id 0..7
  const int lane = tid & 31;
  const int r    = lane & 15;
  const int g    = lane >> 4;
  const int row0 = blockIdx.x * 16;

  // ---- Stage 1: stage the 16x512 fp32 x-tile into LDS, then convert to
  // bf16 (x - bias_pre). The fp32 staging buffer aliases the (not yet live)
  // logits region, so no extra LDS is consumed.
  float* Xstage = Llog;            // 16*512*4 = 32KB, contiguous (no pitch)

#if __has_builtin(__builtin_amdgcn_tensor_load_to_lds)
  if (w == 0) {
    // 2D Tensor DMA descriptor (D#): data_size=4B, tile 512x16, row stride 512.
    uint64_t ga = (uint64_t)(uintptr_t)(x + (size_t)row0 * H_);
    u32x4 g0 = {};
    g0[0] = 1u;                                              // count=1, user D#
    g0[1] = (uint32_t)(uintptr_t)Xstage;                     // lds_addr (addr[31:0])
    g0[2] = (uint32_t)ga;                                    // global_addr lo
    g0[3] = (uint32_t)((ga >> 32) & 0x1FFFFFFu) | (2u << 30); // global_addr hi | type=2
    i32x8 g1 = {};
    g1[0] = (int)(2u << 16);     // workgroup_mask=0 | data_size=2 (4 bytes)
    g1[1] = (int)(512u << 16);   // tensor_dim0[15:0]=512 at bits 63:48
    g1[2] = (int)(16u << 16);    // tensor_dim0 hi=0 | tensor_dim1=16 at bits 111:80
    g1[3] = (int)(512u << 16);   // tile_dim0=512 at bits 127:112
    g1[4] = 16;                  // tile_dim1=16
    g1[5] = 512;                 // tensor_dim0_stride=512 (data_size units)
    i32x4 gz = {};
#if __has_include(<hip/amd_detail/amd_gfx1250_TDM.h>)
    i32x8 gz8 = {};
    __builtin_amdgcn_tensor_load_to_lds(g0, g1, gz, gz, gz8, 0);   // clang-23 arity
#else
    __builtin_amdgcn_tensor_load_to_lds(g0, g1, gz, gz, 0);        // ROCm 7.2 arity
#endif
    __builtin_amdgcn_s_wait_tensorcnt(0);
  }
  __syncthreads();
  for (int i = tid; i < 16 * H_; i += 256) {
    int rr = i >> 9, h = i & (H_ - 1);
    As[rr * A_PITCH + h] = f32_to_bf16(Xstage[i] - bias_pre[h]);
  }
#else
  for (int i = tid; i < 16 * H_; i += 256) {
    int rr = i >> 9, h = i & (H_ - 1);
    float v = x[(size_t)(row0 + rr) * H_ + h] - bias_pre[h];
    As[rr * A_PITCH + h] = f32_to_bf16(v);
  }
#endif
  __syncthreads();

  // ---- Stage 2: WMMA GEMM over all 256 column tiles (K=512 in 16 steps) ----
  for (int t = w; t < M_ / 16; t += 8) {
    const int c0 = t * 16;
    if (t + 8 < M_ / 16)
      __builtin_prefetch(encb + (size_t)(c0 + 128 + r) * H_, 0, 1);  // global_prefetch_b8
    v8f acc = {};
    const __bf16* brow = encb + (size_t)(c0 + r) * H_;
#pragma unroll
    for (int kk = 0; kk < 16; ++kk) {
      const int kb = kk * 32;
      // A fragment (16-bit A 16x32): lane(r,g), element e -> K = kb + g*16 + e (contiguous, matches B)
      v8bf a0 = *(const v8bf*)&As[r * A_PITCH + kb + g * 16];
      v8bf a1 = *(const v8bf*)&As[r * A_PITCH + kb + g * 16 + 8];
      v16bf a = __builtin_shufflevector(a0, a1, 0,1,2,3,4,5,6,7,8,9,10,11,12,13,14,15);
      // B fragment: column n = c0+r (enc row), K = kb + g*16 + e (contiguous, 32B aligned)
      v16bf b = *(const v16bf*)&brow[kb + g * 16];
      acc = __builtin_amdgcn_wmma_f32_16x16x32_bf16(false, a, false, b,
                                                    (short)0, acc, false, false);
    }
    // C layout: lanes = column c0+r; VGPR vv = row vv + 8*g. Bias + ReLU -> LDS.
    const float be = bias_enc[c0 + r];
#pragma unroll
    for (int vv = 0; vv < 8; ++vv) {
      float lv = acc[vv] + be;
      lv = lv > 0.0f ? lv : 0.0f;
      Llog[(vv + g * 8) * L_PITCH + (c0 + r)] = lv;
    }
  }
  __syncthreads();

  // ---- Stage 3: exact top-64 per row via 4-level radix select on fp32 bits
  // (values >= 0 post-ReLU => uint order == float order). One wave per 2 rows.
  unsigned* hist = Hist + w * 256;
  for (int rr = 2 * w; rr < 2 * w + 2; ++rr) {
    const float* rowp = Llog + rr * L_PITCH;
    unsigned T = 0;
    int need = TOPK_;
    for (int level = 3; level >= 0; --level) {
      for (int bq = lane; bq < 256; bq += 32) hist[bq] = 0u;
      const unsigned hiMask = (level == 3) ? 0u : (0xFFFFFFFFu << ((level + 1) * 8));
      for (int i = 0; i < 128; ++i) {
        unsigned u = __float_as_uint(rowp[lane + 32 * i]);
        if ((u & hiMask) == T)
          atomicAdd(&hist[(u >> (level * 8)) & 255u], 1u);
      }
      unsigned cum = 0; int sel = 0;
      for (int bq = 255; bq >= 0; --bq) {        // redundant per-lane scan (cheap, uniform)
        unsigned c = hist[bq];
        if (cum + c >= (unsigned)need) { sel = bq; break; }
        cum += c;
      }
      T |= ((unsigned)sel) << (level * 8);
      need -= (int)cum;                          // still needed among ties at this byte
    }
    // T = bits of 64th-largest value; take all > T plus `need` values == T.
    if (lane == 0) { Cnt[w] = need; Cnt[8 + w] = 0; }
    float* zrow = z_out + (size_t)(row0 + rr) * M_;
    float* tv   = topv  + (size_t)(row0 + rr) * TOPK_;
    int*   ti   = topi  + (size_t)(row0 + rr) * TOPK_;
    for (int i = 0; i < 128; ++i) {
      int j = lane + 32 * i;
      float v = rowp[j];
      unsigned u = __float_as_uint(v);
      bool take = (u > T);
      if (!take && u == T) {
        int old = atomicSub(&Cnt[w], 1);
        take = (old > 0);
      }
      __builtin_nontemporal_store(take ? v : 0.0f, &zrow[j]);   // 268MB streaming output
      if (take) {
        int s = atomicAdd(&Cnt[8 + w], 1);
        if (s < TOPK_) { tv[s] = v; ti[s] = j; }
      }
    }
  }
}

// ---------------------------------------------------------------------------
// Kernel 4: sparse decode: x_tgt[n,:] = bias_pre + sum_j val_j * dec_t[idx_j,:]
// ---------------------------------------------------------------------------
__global__ __launch_bounds__(128)
void sae_decode(const float* __restrict__ topv, const int* __restrict__ topi,
                const float* __restrict__ dect, const float* __restrict__ bias_pre,
                float* __restrict__ xt) {
  __shared__ float sv[TOPK_];
  __shared__ int   si[TOPK_];
  const int row = blockIdx.x;
  const int tid = threadIdx.x;
  if (tid < TOPK_) {
    sv[tid] = topv[(size_t)row * TOPK_ + tid];
    si[tid] = topi[(size_t)row * TOPK_ + tid];
  }
  __syncthreads();
  const int h4 = tid;                        // 128 float4 chunks = 512 floats
  float4 acc = ((const float4*)bias_pre)[h4];
#pragma unroll 4
  for (int j = 0; j < TOPK_; ++j) {
    float  v = sv[j];
    float4 d = ((const float4*)(dect + (size_t)si[j] * H_))[h4];
    acc.x = fmaf(v, d.x, acc.x);
    acc.y = fmaf(v, d.y, acc.y);
    acc.z = fmaf(v, d.z, acc.z);
    acc.w = fmaf(v, d.w, acc.w);
  }
  ((float4*)(xt + (size_t)row * H_))[h4] = acc;
}

// ---------------------------------------------------------------------------
extern "C" void kernel_launch(void* const* d_in, const int* in_sizes, int n_in,
                              void* d_out, int out_size, void* d_ws, size_t ws_size,
                              hipStream_t stream) {
  (void)in_sizes; (void)n_in; (void)out_size; (void)ws_size;
  const float* zL   = (const float*)d_in[0];   // [16384, 512]
  const float* enc  = (const float*)d_in[1];   // [4096, 512]
  const float* dec  = (const float*)d_in[2];   // [512, 4096]
  const float* bpre = (const float*)d_in[3];   // [512]
  const float* benc = (const float*)d_in[4];   // [4096]

  float* z_out  = (float*)d_out;                       // [16384, 4096]
  float* xt_out = z_out + (size_t)NROWS * M_;          // [16384, 512]

  char*   ws   = (char*)d_ws;
  float*  dect = (float*)ws;                                             // 8 MiB
  __bf16* encb = (__bf16*)(ws + (size_t)H_ * M_ * 4);                    // 4 MiB
  float*  topv = (float*)(ws + (size_t)H_ * M_ * 4 + (size_t)M_ * H_ * 2); // 4 MiB
  int*    topi = (int*)((char*)topv + (size_t)NROWS * TOPK_ * 4);          // 4 MiB

  hipFuncSetAttribute((const void*)sae_encode_topk,
                      hipFuncAttributeMaxDynamicSharedMemorySize, SMEM_BYTES);

  sae_transpose_dec<<<dim3(M_ / 32, H_ / 32), dim3(32, 8), 0, stream>>>(dec, dect);
  sae_convert_enc<<<(M_ * H_) / 256, 256, 0, stream>>>(enc, encb);
  sae_encode_topk<<<NROWS / 16, 256, SMEM_BYTES, stream>>>(zL, encb, bpre, benc,
                                                           z_out, topv, topi);
  sae_decode<<<NROWS, 128, 0, stream>>>(topv, topi, dect, bpre, xt_out);
}
